// MDTA_47820165874436
// MI455X (gfx1250) — matrix operand order repaired and checked by reading, not compiled, not run on hardware
//
#include <hip/hip_runtime.h>
#include <math.h>

// ---------------- problem constants (from reference) ----------------
#define BATCH 8
#define CCH   192          // C
#define C3    576          // 3C
#define HEADS 4
#define HD    48           // C / HEADS
#define HH    128
#define WW    128
#define NPIX  16384        // H*W
#define EPS   1e-12f
#define KSPLIT 64          // split-K factor for Gram kernel (deterministic)

// per-split partial block: 48x48 Gram for 32 (b,h) pairs + sumsq(q)+sumsq(k)
#define GSZ   (32 * HD * HD)          // 73728
#define SSZ   (2 * BATCH * CCH)       // 3072
#define PART  (GSZ + SSZ)             // 76800

typedef __attribute__((ext_vector_type(16))) _Float16 v16h;
typedef __attribute__((ext_vector_type(8)))  float    v8f;

// ---- WMMA f16 16x16x32 fragment index helpers (CDNA5 ISA 7.12.2) ----
// A (16x32, MxK): lanes 0-15 hold M=lane, K = e; lanes 16-31 K = 16+e (this code's order, not the ISA table).
__device__ __forceinline__ int a_koff(int lane, int e) {
    return ((lane >> 4) ? 16 : 0) + e;
}
// B (32x16, KxN): lanes 0-15 hold N=lane, K = e; lanes 16-31 K = 16+e (same order as A, not the ISA table).
__device__ __forceinline__ int b_koff(int lane, int e) {
    return ((lane >> 4) ? 16 : 0) + e;
}
// C/D (16x16 f32): lane 0-15 -> N=lane, M=v; lane 16-31 -> N=lane-16, M=8+v.

// ---------------- async global->LDS (gfx1250 ASYNCcnt path) ----------
#if defined(__has_builtin)
#  if __has_builtin(__builtin_amdgcn_global_load_async_to_lds_b32)
#    define HAVE_ASYNC_LDS 1
#  endif
#endif

typedef __attribute__((address_space(1))) int glb_i32;
typedef __attribute__((address_space(3))) int lds_i32;

__device__ __forceinline__ void async_copy_f32(float* lds_dst, const float* gsrc) {
#ifdef HAVE_ASYNC_LDS
    __builtin_amdgcn_global_load_async_to_lds_b32(
        (glb_i32*)(float*)gsrc, (lds_i32*)lds_dst, 0, 0);
#else
    *lds_dst = *gsrc;
#endif
}
__device__ __forceinline__ void async_wait_all() {
#ifdef HAVE_ASYNC_LDS
#  if __has_builtin(__builtin_amdgcn_s_wait_asynccnt)
    __builtin_amdgcn_s_wait_asynccnt(0);
#  else
    asm volatile("s_wait_asynccnt 0x0" ::: "memory");
#  endif
#endif
}

// --------------------------------------------------------------------
// Generic batched GEMM: C[z] = A[z] (M x K, row major) * B[z] (K x N, row major)
// fp32 in/out, f16 WMMA compute, f32 accumulate.
// block = (32, WAVES); each wave computes a 16 x 64 tile of C.
// grid  = (N/64, M/(16*WAVES), batch)
// --------------------------------------------------------------------
__global__ void gemm_f16_wmma(const float* __restrict__ A, long aStride,
                              const float* __restrict__ B, long bStride,
                              float* __restrict__ C, long cStride,
                              int M, int N, int K) {
    const int lane  = threadIdx.x;             // 0..31
    const int wave  = threadIdx.y;
    const int WAVES = blockDim.y;
    const long z    = blockIdx.z;
    const float* Ab = A + z * aStride;
    const float* Bb = B + z * bStride;
    float*       Cb = C + z * cStride;

    const int m0 = (blockIdx.y * WAVES + wave) * 16;
    const int n0 = blockIdx.x * 64;

    v8f acc[4] = {v8f{}, v8f{}, v8f{}, v8f{}};

    const int arow = m0 + (lane & 15);
    const int bcol = lane & 15;

    for (int k0 = 0; k0 < K; k0 += 32) {
        if (k0 + 32 < K) {  // prefetch next K tile -> global_prefetch_b8
            __builtin_prefetch(&Ab[(long)arow * K + (k0 + 32) + (lane & 31)], 0, 3);
            __builtin_prefetch(&Bb[(long)(k0 + 32 + (lane >> 4) * 16) * N + n0 + bcol], 0, 3);
        }
        v16h afr;
#pragma unroll
        for (int e = 0; e < 16; ++e)
            afr[e] = (_Float16)Ab[(long)arow * K + (k0 + a_koff(lane, e))];
#pragma unroll
        for (int j = 0; j < 4; ++j) {
            v16h bfr;
#pragma unroll
            for (int e = 0; e < 16; ++e)
                bfr[e] = (_Float16)Bb[(long)(k0 + b_koff(lane, e)) * N + (n0 + j * 16 + bcol)];
            acc[j] = __builtin_amdgcn_wmma_f32_16x16x32_f16(
                false, afr, false, bfr, (short)0, acc[j], false, false);
        }
    }

    const int mrow = m0 + ((lane >> 4) << 3);
#pragma unroll
    for (int j = 0; j < 4; ++j)
#pragma unroll
        for (int v = 0; v < 8; ++v)
            Cb[(long)(mrow + v) * N + (n0 + j * 16 + bcol)] = acc[j][v];
}

// --------------------------------------------------------------------
// Depthwise 3x3, padding=1, groups=3C.
// One block = one (b,ch) plane strip of 8 output rows; 10-row halo tile
// staged in LDS via async global->LDS loads (ASYNCcnt), zero-padded edges.
// grid = B*C3*16 blocks of 256 threads.
// --------------------------------------------------------------------
__global__ void dwconv3x3(const float* __restrict__ in,
                          const float* __restrict__ wdw,
                          float* __restrict__ out) {
    const int  blk    = blockIdx.x;
    const int  ystrip = blk & 15;              // 16 strips of 8 rows
    const long bc     = blk >> 4;              // b*576 + ch
    const int  ch     = (int)(bc % C3);
    const int  y0     = ystrip * 8;
    const float* p    = in + (bc << 14);

    __shared__ float tile[10 * WW];            // rows y0-1 .. y0+8

    for (int i = threadIdx.x; i < 10 * WW; i += 256) {
        const int r  = i >> 7;
        const int xx = i & (WW - 1);
        const int ys = y0 - 1 + r;
        if (ys >= 0 && ys < HH)
            async_copy_f32(&tile[i], &p[((long)ys << 7) + xx]);
        else
            tile[i] = 0.f;
    }
    async_wait_all();
    __syncthreads();

    float wk[9];
#pragma unroll
    for (int t = 0; t < 9; ++t) wk[t] = wdw[(long)ch * 9 + t];

    for (int i = threadIdx.x; i < 8 * WW; i += 256) {
        const int r  = i >> 7;                 // 0..7 (output row y0+r)
        const int xx = i & (WW - 1);
        float s = 0.f;
#pragma unroll
        for (int dy = 0; dy < 3; ++dy) {
#pragma unroll
            for (int dx = -1; dx <= 1; ++dx) {
                const int xs = xx + dx;
                if (xs < 0 || xs >= WW) continue;
                s += wk[dy * 3 + (dx + 1)] * tile[(r + dy) * WW + xs];
            }
        }
        out[(bc << 14) + (((long)(y0 + r)) << 7) + xx] = s;
    }
}

// --------------------------------------------------------------------
// Gram partials + fused sum-of-squares partials (deterministic split-K):
//   Part[split] = { G[32][48][48] , Sq[1536] , Sk[1536] }
// block = (32, 9): 9 waves tile the 48x48 output as 3x3 WMMA tiles.
// grid  = (KSPLIT, 32).
// --------------------------------------------------------------------
__global__ void gram_wmma(const float* __restrict__ dwqkv, float* __restrict__ Part) {
    const int lane = threadIdx.x;
    const int wave = threadIdx.y;              // 0..8
    const int mt = wave / 3, nt = wave % 3;
    const int bh = blockIdx.y;                 // 0..31
    const int b = bh / HEADS, h = bh % HEADS;
    const int  kspan = NPIX / KSPLIT;          // 256
    const long kbase = (long)blockIdx.x * kspan;

    const float* q  = dwqkv + ((long)b * C3 + (long)h * HD) * NPIX;
    const float* km = dwqkv + ((long)b * C3 + CCH + (long)h * HD) * NPIX;

    const int arow = mt * 16 + (lane & 15);    // q row (c)
    const int bcol = nt * 16 + (lane & 15);    // k row (d) == B column

    v8f acc = {};
    float sq = 0.f, sk = 0.f;
    for (int k0 = 0; k0 < kspan; k0 += 32) {
        v16h afr, bfr;
#pragma unroll
        for (int e = 0; e < 16; ++e) {
            const float v = q[(long)arow * NPIX + kbase + k0 + a_koff(lane, e)];
            sq += v * v;
            afr[e] = (_Float16)v;
        }
#pragma unroll
        for (int e = 0; e < 16; ++e) {
            const float v = km[(long)bcol * NPIX + kbase + k0 + b_koff(lane, e)];
            sk += v * v;
            bfr[e] = (_Float16)v;
        }
        acc = __builtin_amdgcn_wmma_f32_16x16x32_f16(
            false, afr, false, bfr, (short)0, acc, false, false);
    }

    float* Pp = Part + (long)blockIdx.x * PART;

    // Gram tile
    const int m = mt * 16 + ((lane >> 4) << 3);
    const int n = nt * 16 + (lane & 15);
    float* Gp = Pp + (long)bh * HD * HD;
#pragma unroll
    for (int v = 0; v < 8; ++v)
        Gp[(long)(m + v) * HD + n] = acc[v];

    // sum-of-squares: lanes l and l+16 together cover the full k-chunk of a row
    sq += __shfl_xor(sq, 16);
    sk += __shfl_xor(sk, 16);
    if (nt == 0 && lane < 16)                      // waves 0,3,6 -> q rows 0..47
        Pp[GSZ + (long)bh * HD + mt * 16 + lane] = sq;
    if (mt == 0 && lane < 16)                      // waves 0,1,2 -> k rows 0..47
        Pp[GSZ + 1536 + (long)bh * HD + nt * 16 + lane] = sk;
}

// Fixed-order reduction over splits -> deterministic {G, S}.
__global__ void reduce_parts(const float* __restrict__ Part, float* __restrict__ R) {
    const long idx = (long)blockIdx.x * blockDim.x + threadIdx.x;
    if (idx >= PART) return;
    float s = 0.f;
    for (int sp = 0; sp < KSPLIT; ++sp)
        s += Part[(long)sp * PART + idx];
    R[idx] = s;
}

// --------------------------------------------------------------------
// attn = softmax_row( G * temp / (max(||q_c||,eps) * max(||k_d||,eps)) )
// one block (64 thr) per (b,h,c) row of 48.  S layout: [qk][b][h][c]
// --------------------------------------------------------------------
__global__ void attn_softmax(const float* __restrict__ G, const float* __restrict__ S,
                             const float* __restrict__ temp, float* __restrict__ attn) {
    const int row = blockIdx.x;                // 0..1535
    const int bh = row / HD, c = row % HD;
    const int d = threadIdx.x;                 // 0..63
    const float t  = temp[0];
    const float nq = fmaxf(sqrtf(S[bh * HD + c]), EPS);

    __shared__ float buf[64];
    float val = -1e30f;
    if (d < HD) {
        const float nk = fmaxf(sqrtf(S[1536 + bh * HD + d]), EPS);
        val = G[((long)bh * HD + c) * HD + d] * t / (nq * nk);
    }
    buf[d] = val;
    __syncthreads();
    for (int off = 32; off; off >>= 1) {
        if (d < off) buf[d] = fmaxf(buf[d], buf[d + off]);
        __syncthreads();
    }
    const float mx = buf[0];
    __syncthreads();
    const float e = (d < HD) ? __expf(val - mx) : 0.f;
    buf[d] = e;
    __syncthreads();
    for (int off = 32; off; off >>= 1) {
        if (d < off) buf[d] += buf[d + off];
        __syncthreads();
    }
    if (d < HD) attn[((long)bh * HD + c) * HD + d] = e / buf[0];
}

// --------------------------------------------------------------------
// Fold proj into attention:  Mm[b][o][kf=h*48+d] = sum_c wproj[o,h*48+c]*attn[b,h,c,d]
// --------------------------------------------------------------------
__global__ void build_M(const float* __restrict__ wproj, const float* __restrict__ attn,
                        float* __restrict__ Mm) {
    const long idx = (long)blockIdx.x * blockDim.x + threadIdx.x;
    const long total = (long)BATCH * CCH * CCH;
    if (idx >= total) return;
    const int kf = (int)(idx % CCH);
    const int o  = (int)((idx / CCH) % CCH);
    const int b  = (int)(idx / (CCH * CCH));
    const int h = kf / HD, d = kf % HD;
    const float* wrow = wproj + (long)o * CCH + h * HD;
    const float* acol = attn + (((long)b * HEADS + h) * HD) * HD + d; // stride HD over c
    float s = 0.f;
#pragma unroll 4
    for (int c = 0; c < HD; ++c) s += wrow[c] * acol[(long)c * HD];
    Mm[idx] = s;
}

// --------------------------------------------------------------------
extern "C" void kernel_launch(void* const* d_in, const int* in_sizes, int n_in,
                              void* d_out, int out_size, void* d_ws, size_t ws_size,
                              hipStream_t stream) {
    (void)in_sizes; (void)n_in; (void)out_size; (void)ws_size;
    const float* x      = (const float*)d_in[0];   // [8,192,128,128]
    const float* w_pw   = (const float*)d_in[1];   // [576,192]
    const float* w_dw   = (const float*)d_in[2];   // [576,1,3,3]
    const float* w_proj = (const float*)d_in[3];   // [192,192]
    const float* temp   = (const float*)d_in[4];   // scalar
    float*       out    = (float*)d_out;           // [8,192,128,128]

    const long QKV = (long)BATCH * C3 * NPIX;      // 75,497,472 floats
    float* qkv   = (float*)d_ws;                   // pw result
    float* dwqkv = qkv + QKV;                      // depthwise result
    float* Part  = dwqkv + QKV;                    // KSPLIT * PART
    float* R     = Part + (long)KSPLIT * PART;     // {G[73728], S[3072]}
    float* attn  = R + PART;                       // 73728
    float* Mm    = attn + (long)GSZ;               // [8,192,192]

    // 1) qkv = w_pw @ x   (per batch: 576 x 16384, K=192)
    gemm_f16_wmma<<<dim3(NPIX / 64, C3 / 96, BATCH), dim3(32, 6), 0, stream>>>(
        w_pw, 0L, x, (long)CCH * NPIX, qkv, (long)C3 * NPIX, C3, NPIX, CCH);

    // 2) depthwise 3x3 (async global->LDS staged halo tiles)
    dwconv3x3<<<BATCH * C3 * 16, 256, 0, stream>>>(qkv, w_dw, dwqkv);

    // 3) Gram partials + fused sumsq, then deterministic reduction
    gram_wmma<<<dim3(KSPLIT, 32), dim3(32, 9), 0, stream>>>(dwqkv, Part);
    reduce_parts<<<(PART + 255) / 256, 256, 0, stream>>>(Part, R);

    // 4) normalize + temperature + softmax
    attn_softmax<<<1536, 64, 0, stream>>>(R, R + GSZ, temp, attn);

    // 5) Mm[b] = w_proj @ blockdiag(attn[b])
    build_M<<<(int)(((long)BATCH * CCH * CCH + 255) / 256), 256, 0, stream>>>(
        w_proj, attn, Mm);

    // 6) out = Mm[b] @ v   (v = dwqkv channels [384..576))
    gemm_f16_wmma<<<dim3(NPIX / 64, CCH / 96, BATCH), dim3(32, 6), 0, stream>>>(
        Mm, (long)CCH * CCH,
        dwqkv + (long)2 * CCH * NPIX, (long)C3 * NPIX,
        out, (long)CCH * NPIX,
        CCH, NPIX, CCH);
}
